// MSDeformAttnAdapter_25761213841553
// MI455X (gfx1250) — matrix operand order repaired and checked by reading, not compiled, not run on hardware
//
#include <hip/hip_runtime.h>
#include <hip/hip_bf16.h>

typedef __attribute__((ext_vector_type(16))) _Float16 v16h;
typedef __attribute__((ext_vector_type(8)))  _Float16 v8h;
typedef __attribute__((ext_vector_type(8)))  float    v8f;

#define B_    2
#define LQ_   12240
#define DM_   256
#define MROWS (B_ * LQ_)   // 24480 rows for all GEMMs

// ---------------------------------------------------------------------------
// f32 -> f16 cast
// ---------------------------------------------------------------------------
__global__ void k_cast_half(const float* __restrict__ src, _Float16* __restrict__ dst, int n) {
  int i = blockIdx.x * blockDim.x + threadIdx.x;
  if (i < n) dst[i] = (_Float16)src[i];
}

// ---------------------------------------------------------------------------
// Weight transpose to [N][K] f16 so each WMMA lane's B fragment (a K-column)
// is one contiguous 32B load.
// ---------------------------------------------------------------------------
__global__ void k_transpose_half(const float* __restrict__ W, _Float16* __restrict__ Wt,
                                 int K, int N) {
  int i = blockIdx.x * blockDim.x + threadIdx.x;
  if (i >= K * N) return;
  int n = i / K;
  int k = i - n * K;
  Wt[i] = (_Float16)W[(size_t)k * N + n];
}

// ---------------------------------------------------------------------------
// Build concatenated multi-scale value matrix [B*LQ, 256] (f16) from the
// channel-major feature maps [B,256,H,W].
// ---------------------------------------------------------------------------
__global__ void k_build_value(const float* __restrict__ f0, const float* __restrict__ f1,
                              const float* __restrict__ f2, const float* __restrict__ f3,
                              const int* __restrict__ ss, _Float16* __restrict__ vh) {
  int i = blockIdx.x * blockDim.x + threadIdx.x;
  if (i >= MROWS * DM_) return;
  int c   = i & (DM_ - 1);
  int t   = i >> 8;
  int pos = t % LQ_;
  int b   = t / LQ_;
  int s0 = ss[0] * ss[1];
  int s1 = ss[2] * ss[3];
  int s2 = ss[4] * ss[5];
  int s3 = ss[6] * ss[7];
  const float* f; int local, hw;
  if      (pos < s0)           { f = f0; local = pos;                hw = s0; }
  else if (pos < s0 + s1)      { f = f1; local = pos - s0;           hw = s1; }
  else if (pos < s0 + s1 + s2) { f = f2; local = pos - s0 - s1;      hw = s2; }
  else                         { f = f3; local = pos - s0 - s1 - s2; hw = s3; }
  vh[i] = (_Float16)f[((size_t)b * DM_ + c) * hw + local];
}

// ---------------------------------------------------------------------------
// WMMA GEMM: C[M,N] = A[M,K] (f16, row-major) * Bt[N,K] (f16, pre-transposed)
//            + bias[N], outputs f32 (Cf) and/or f16 (Ch).
// One wave per 16x64 C tile: 4 accumulators share a single A fragment.
// K and M multiples of 32/16; N multiple of 64 (256 or 128 here).
// ---------------------------------------------------------------------------
__global__ void __launch_bounds__(256)
k_gemm_wmma(const _Float16* __restrict__ A, const _Float16* __restrict__ Bt,
            const float* __restrict__ bias, float* __restrict__ Cf,
            _Float16* __restrict__ Ch, int M, int N, int K) {
  int wave = (blockIdx.x * blockDim.x + threadIdx.x) >> 5;
  int lane = threadIdx.x & 31;
  int Nt = N >> 6;
  int Mt = M >> 4;
  if (wave >= Mt * Nt) return;
  int tm = wave / Nt, tn = wave - tm * Nt;
  int row0 = tm << 4, col0 = tn << 6;

  int ln = lane & 15;        // row of A tile / column within 16-wide B tile
  int hi = lane >> 4;        // half-wave select
  // A fragment: lane ln holds row (row0+ln); halves 0..15 = K[16*hi .. +15]
  // contiguous (same k order as B)
  const _Float16* Arow = A + (size_t)(row0 + ln) * K + (hi << 4);
  // B fragment: lane holds column (col0+sub*16+ln), K[16*hi .. +15] contiguous in Bt
  const _Float16* Bc0 = Bt + (size_t)(col0 +  0 + ln) * K + (hi << 4);
  const _Float16* Bc1 = Bt + (size_t)(col0 + 16 + ln) * K + (hi << 4);
  const _Float16* Bc2 = Bt + (size_t)(col0 + 32 + ln) * K + (hi << 4);
  const _Float16* Bc3 = Bt + (size_t)(col0 + 48 + ln) * K + (hi << 4);

  v8f c0 = {}, c1 = {}, c2 = {}, c3 = {};
  for (int kt = 0; kt < K; kt += 32) {
    union { v16h v; v8h h[2]; } ua;
    ua.h[0] = *(const v8h*)(Arow + kt);
    ua.h[1] = *(const v8h*)(Arow + kt + 8);
    v16h a  = ua.v;
    v16h b0 = *(const v16h*)(Bc0 + kt);
    v16h b1 = *(const v16h*)(Bc1 + kt);
    v16h b2 = *(const v16h*)(Bc2 + kt);
    v16h b3 = *(const v16h*)(Bc3 + kt);
    c0 = __builtin_amdgcn_wmma_f32_16x16x32_f16(false, a, false, b0, (short)0, c0, false, false);
    c1 = __builtin_amdgcn_wmma_f32_16x16x32_f16(false, a, false, b1, (short)0, c1, false, false);
    c2 = __builtin_amdgcn_wmma_f32_16x16x32_f16(false, a, false, b2, (short)0, c2, false, false);
    c3 = __builtin_amdgcn_wmma_f32_16x16x32_f16(false, a, false, b3, (short)0, c3, false, false);
  }

  float bb0 = bias[col0 + ln], bb1 = bias[col0 + 16 + ln];
  float bb2 = bias[col0 + 32 + ln], bb3 = bias[col0 + 48 + ln];
  #pragma unroll
  for (int r = 0; r < 8; ++r) {
    int row = row0 + r + (hi << 3);                 // C layout: VGPR r -> M=r / r+8
    size_t base = (size_t)row * N + col0 + ln;
    float x0 = c0[r] + bb0, x1 = c1[r] + bb1, x2 = c2[r] + bb2, x3 = c3[r] + bb3;
    if (Cf) { Cf[base] = x0; Cf[base + 16] = x1; Cf[base + 32] = x2; Cf[base + 48] = x3; }
    if (Ch) {
      Ch[base]      = (_Float16)x0; Ch[base + 16] = (_Float16)x1;
      Ch[base + 32] = (_Float16)x2; Ch[base + 48] = (_Float16)x3;
    }
  }
}

// ---------------------------------------------------------------------------
// Softmax over 16 attention logits per (b,q,h), in place.
// ---------------------------------------------------------------------------
__global__ void k_softmax16(float* __restrict__ aw, int rows) {
  int r = blockIdx.x * blockDim.x + threadIdx.x;
  if (r >= rows) return;
  float* p = aw + (size_t)r * 16;
  float m = p[0];
  #pragma unroll
  for (int i = 1; i < 16; ++i) m = fmaxf(m, p[i]);
  float e[16], s = 0.f;
  #pragma unroll
  for (int i = 0; i < 16; ++i) { e[i] = __expf(p[i] - m); s += e[i]; }
  float inv = 1.f / s;
  #pragma unroll
  for (int i = 0; i < 16; ++i) p[i] = e[i] * inv;
}

// ---------------------------------------------------------------------------
// Deformable sampling: one wave per (b,q,h); lane = head-dim element (HEAD_DIM
// == 32 == wave32). Each corner fetch is one coalesced 128B load from L2.
// ---------------------------------------------------------------------------
__global__ void __launch_bounds__(256)
k_sample(const float* __restrict__ v, const float* __restrict__ off,
         const float* __restrict__ aw, const float* __restrict__ ref,
         const int* __restrict__ ss, const int* __restrict__ lsi,
         _Float16* __restrict__ out) {
  int gw   = (blockIdx.x * blockDim.x + threadIdx.x) >> 5;
  int lane = threadIdx.x & 31;
  if (gw >= MROWS * 8) return;
  int h  = gw & 7;
  int bq = gw >> 3;            // b*LQ + q
  int b  = bq / LQ_;

  const float* refp = ref + (size_t)bq * 8;               // [4 levels][2]
  const float* offp = off + (size_t)bq * DM_ + h * 32;    // [L][P][2] for this head
  const float* awp  = aw  + (size_t)bq * 128 + h * 16;    // [L][P]
  const float* vb   = v + ((size_t)b * LQ_) * DM_ + h * 32 + lane;

  float acc = 0.f;
  #pragma unroll
  for (int l = 0; l < 4; ++l) {
    int   Hl = ss[2 * l], Wl = ss[2 * l + 1];
    int   st = lsi[l];
    float fw = (float)Wl, fh = (float)Hl;
    float rx = refp[2 * l], ry = refp[2 * l + 1];
    #pragma unroll
    for (int p = 0; p < 4; ++p) {
      float ox = offp[l * 8 + p * 2];
      float oy = offp[l * 8 + p * 2 + 1];
      float a_ = awp[l * 4 + p];
      float x  = (rx + ox / fw) * fw - 0.5f;
      float y  = (ry + oy / fh) * fh - 0.5f;
      float x0 = floorf(x), y0 = floorf(y);
      #pragma unroll
      for (int dy = 0; dy < 2; ++dy) {
        #pragma unroll
        for (int dx = 0; dx < 2; ++dx) {
          float xi = x0 + (float)dx, yi = y0 + (float)dy;
          if (xi >= 0.f && xi < fw && yi >= 0.f && yi < fh) {
            float wgt = (1.f - fabsf(x - xi)) * (1.f - fabsf(y - yi));
            int idx = st + (int)yi * Wl + (int)xi;
            acc += a_ * wgt * vb[(size_t)idx * DM_];
          }
        }
      }
    }
  }
  out[(size_t)bq * DM_ + h * 32 + lane] = (_Float16)acc;
}

// ---------------------------------------------------------------------------
extern "C" void kernel_launch(void* const* d_in, const int* in_sizes, int n_in,
                              void* d_out, int out_size, void* d_ws, size_t ws_size,
                              hipStream_t stream) {
  (void)in_sizes; (void)n_in; (void)out_size; (void)ws_size;
  const float* q     = (const float*)d_in[0];
  const float* ref   = (const float*)d_in[1];
  const int*   ss    = (const int*)d_in[2];
  const int*   lsi   = (const int*)d_in[3];
  const float* f0    = (const float*)d_in[4];
  const float* f1    = (const float*)d_in[5];
  const float* f2    = (const float*)d_in[6];
  const float* f3    = (const float*)d_in[7];
  const float* Wq    = (const float*)d_in[8];
  const float* bq    = (const float*)d_in[9];
  const float* Woff  = (const float*)d_in[10];
  const float* boff  = (const float*)d_in[11];
  const float* Wattn = (const float*)d_in[12];
  const float* battn = (const float*)d_in[13];
  const float* Wval  = (const float*)d_in[14];
  const float* bval  = (const float*)d_in[15];
  const float* Wproj = (const float*)d_in[16];
  const float* bproj = (const float*)d_in[17];
  const float* Wout  = (const float*)d_in[18];
  const float* bout  = (const float*)d_in[19];

  char* wp = (char*)d_ws;
  auto take = [&](size_t bytes) { char* p = wp; wp += (bytes + 255) & ~(size_t)255; return p; };
  _Float16* qh     = (_Float16*)take((size_t)MROWS * DM_ * 2);
  _Float16* vh     = (_Float16*)take((size_t)MROWS * DM_ * 2);
  _Float16* qph    = (_Float16*)take((size_t)MROWS * DM_ * 2);
  _Float16* samph  = (_Float16*)take((size_t)MROWS * DM_ * 2);
  _Float16* projh  = (_Float16*)take((size_t)MROWS * DM_ * 2);
  float*    vf     = (float*)take((size_t)MROWS * DM_ * 4);
  float*    offf   = (float*)take((size_t)MROWS * DM_ * 4);
  float*    awf    = (float*)take((size_t)MROWS * 128 * 4);
  _Float16* wqt    = (_Float16*)take(256 * 256 * 2);
  _Float16* wofft  = (_Float16*)take(256 * 256 * 2);
  _Float16* wattnt = (_Float16*)take(128 * 256 * 2);
  _Float16* wvalt  = (_Float16*)take(256 * 256 * 2);
  _Float16* wprojt = (_Float16*)take(256 * 256 * 2);
  _Float16* woutt  = (_Float16*)take(256 * 256 * 2);

  int nQ = MROWS * DM_;
  k_cast_half  <<<(nQ + 255) / 256, 256, 0, stream>>>(q, qh, nQ);
  k_build_value<<<(nQ + 255) / 256, 256, 0, stream>>>(f0, f1, f2, f3, ss, vh);

  k_transpose_half<<<(256 * 256 + 255) / 256, 256, 0, stream>>>(Wq,    wqt,    256, 256);
  k_transpose_half<<<(256 * 256 + 255) / 256, 256, 0, stream>>>(Woff,  wofft,  256, 256);
  k_transpose_half<<<(256 * 128 + 255) / 256, 256, 0, stream>>>(Wattn, wattnt, 256, 128);
  k_transpose_half<<<(256 * 256 + 255) / 256, 256, 0, stream>>>(Wval,  wvalt,  256, 256);
  k_transpose_half<<<(256 * 256 + 255) / 256, 256, 0, stream>>>(Wproj, wprojt, 256, 256);
  k_transpose_half<<<(256 * 256 + 255) / 256, 256, 0, stream>>>(Wout,  woutt,  256, 256);

  auto gemm = [&](const _Float16* A, const _Float16* Bt, const float* bias,
                  float* Cf, _Float16* Ch, int M, int N, int K) {
    int waves   = (M >> 4) * (N >> 6);
    int threads = waves * 32;
    k_gemm_wmma<<<(threads + 255) / 256, 256, 0, stream>>>(A, Bt, bias, Cf, Ch, M, N, K);
  };

  gemm(qh,    wqt,    bq,    nullptr, qph,   MROWS, 256, 256);  // qp (f16 for reuse)
  gemm(vh,    wvalt,  bval,  vf,      nullptr, MROWS, 256, 256); // v  (f32 for gather)
  gemm(qph,   wofft,  boff,  offf,    nullptr, MROWS, 256, 256); // sampling offsets
  gemm(qph,   wattnt, battn, awf,     nullptr, MROWS, 128, 256); // attention logits

  int rows = MROWS * 8;
  k_softmax16<<<(rows + 255) / 256, 256, 0, stream>>>(awf, rows);

  int sthreads = MROWS * 8 * 32;                                 // wave per (b,q,h)
  k_sample<<<(sthreads + 255) / 256, 256, 0, stream>>>(vf, offf, awf, ref, ss, lsi, samph);

  gemm(samph, wprojt, bproj, nullptr, projh, MROWS, 256, 256);   // output_proj
  gemm(projh, woutt,  bout,  (float*)d_out, nullptr, MROWS, 256, 256); // adapter out_proj
}
